// GraphNeuralNetwork_27728308863842
// MI455X (gfx1250) — matrix operand order repaired and checked by reading, not compiled, not run on hardware
//
#include <hip/hip_runtime.h>
#include <hip/hip_bf16.h>
#include <math.h>

#define N_NODES 50000
#define N_EDGES 800000
#define IN_NODE 30
#define IN_EDGE 10
#define IN_GLOB 100
#define D_NODE 64
#define D_EDGE 32
#define HID 256
#define N_LAYERS 4
#define POLICY_DIM 10000

typedef __attribute__((ext_vector_type(16))) __bf16 v16bf;
typedef __attribute__((ext_vector_type(8)))  float  v8f;

// ---------- helpers ----------

__device__ __forceinline__ unsigned short f2bf(float x) {
  unsigned int u = __float_as_uint(x);
  u += 0x7FFFu + ((u >> 16) & 1u);           // round to nearest even
  return (unsigned short)(u >> 16);
}

__device__ __forceinline__ v16bf make_frag(uint4 lo, uint4 hi) {
  union { uint4 q[2]; v16bf v; } u;
  u.q[0] = lo; u.q[1] = hi;
  return u.v;
}

__device__ __forceinline__ v8f wmma_bf16(v16bf a, v16bf b, v8f c) {
  return __builtin_amdgcn_wmma_f32_16x16x32_bf16(
      /*neg_a=*/false, a, /*neg_b=*/false, b,
      /*c_mod=*/(short)0, c, /*reuse_a=*/false, /*reuse_b=*/false);
}

// ---------- setup kernels ----------

// nodes = x_nodes @ node_w + node_b  ->  bf16
__global__ void embed_nodes_kernel(const float* __restrict__ x,
                                   const float* __restrict__ w,
                                   const float* __restrict__ b,
                                   unsigned short* __restrict__ out) {
  int idx = blockIdx.x * blockDim.x + threadIdx.x;
  if (idx >= N_NODES * D_NODE) return;
  int f = idx & (D_NODE - 1);
  int node = idx >> 6;
  float s = b[f];
  #pragma unroll
  for (int k = 0; k < IN_NODE; k++) s += x[node * IN_NODE + k] * w[k * D_NODE + f];
  out[idx] = f2bf(s);
}

// edges = x_edges @ edge_w + edge_b  ->  bf16
__global__ void embed_edges_kernel(const float* __restrict__ x,
                                   const float* __restrict__ w,
                                   const float* __restrict__ b,
                                   unsigned short* __restrict__ out) {
  long long idx = (long long)blockIdx.x * blockDim.x + threadIdx.x;
  if (idx >= (long long)N_EDGES * D_EDGE) return;
  int f = (int)(idx & (D_EDGE - 1));
  long long e = idx >> 5;
  float s = b[f];
  #pragma unroll
  for (int k = 0; k < IN_EDGE; k++) s += x[e * IN_EDGE + k] * w[k * D_EDGE + f];
  out[idx] = f2bf(s);
}

// Pack one [K,N] f32 weight matrix (one layer) into WMMA B-fragment order, bf16.
// packed[((kt*NT + nt)*32 + lane)*16 + i] = W[(kt*32 + (i/8)*16 + (lane/16)*8 + i%8) * N + nt*16 + lane%16]
__global__ void pack_w_kernel(const float* __restrict__ src,
                              unsigned short* __restrict__ dst,
                              int K, int N) {
  int tid = blockIdx.x * blockDim.x + threadIdx.x;
  if (tid >= K * N) return;
  int i    = tid & 15;
  int lane = (tid >> 4) & 31;
  int rest = tid >> 9;
  int NT = N >> 4;
  int nt = rest % NT;
  int kt = rest / NT;
  int k = kt * 32 + (i >> 3) * 16 + (lane >> 4) * 8 + (i & 7);
  int n = nt * 16 + (lane & 15);
  dst[tid] = f2bf(src[(size_t)k * N + n]);
}

__global__ void zero_f32_kernel(float* __restrict__ p, int n) {
  int i = blockIdx.x * blockDim.x + threadIdx.x;
  if (i < n) p[i] = 0.0f;
}

// ---------- edge message kernel (WMMA) ----------
// 128 threads = 4 waves; block handles 64 edges (N_EDGES % 64 == 0).
// GEMM1: [64x160]bf16 x [160x256] -> ReLU -> bf16 LDS
// GEMM2: [64x256]bf16 x [256x64]  -> f32 atomics scatter into agg[col]
__global__ void __launch_bounds__(128) msg_kernel(
    int L,
    const int* __restrict__ eidx,
    const unsigned short* __restrict__ nodes_in,
    const unsigned short* __restrict__ edges_bf,
    const unsigned short* __restrict__ w1p,
    const float* __restrict__ b1,
    const unsigned short* __restrict__ w2p,
    const float* __restrict__ b2,
    float* __restrict__ agg) {
  __shared__ __align__(16) unsigned short sA[64 * 160];
  __shared__ __align__(16) unsigned short sH[64 * HID];
  __shared__ int sCol[64];

  const int t = threadIdx.x;
  const int e = t & 63;
  const int which = t >> 6;  // 0 = src(row), 1 = tgt(col)
  const long long eBase = (long long)blockIdx.x * 64;

  {  // cooperative gather-stage: concat(src 64 | tgt 64 | edge 32) bf16
    int gi = eidx[(size_t)which * N_EDGES + eBase + e];
    const uint4* src = (const uint4*)(nodes_in + (size_t)gi * D_NODE);
    uint4* dst = (uint4*)(sA + e * 160 + which * 64);
    #pragma unroll
    for (int j = 0; j < 8; j++) dst[j] = src[j];
    if (which) {
      sCol[e] = gi;
    } else {
      const uint4* es = (const uint4*)(edges_bf + (size_t)(eBase + e) * D_EDGE);
      uint4* ed = (uint4*)(sA + e * 160 + 128);
      #pragma unroll
      for (int j = 0; j < 4; j++) ed[j] = es[j];
    }
  }
  __syncthreads();

  const int lane = t & 31;
  const int half = lane >> 4;
  const int wave = t >> 5;
  const int m0 = wave * 16;
  const int mrow = m0 + (lane & 15);

  // ---- GEMM1: K = 160 = 5 k-steps, N = 256 = 16 tiles ----
  v16bf a1[5];
  #pragma unroll
  for (int ks = 0; ks < 5; ks++) {
    const unsigned short* rp = sA + mrow * 160 + ks * 32 + 8 * half;
    a1[ks] = make_frag(*(const uint4*)rp, *(const uint4*)(rp + 16));
  }
  const unsigned short* w1L = w1p + (size_t)L * (160 * 256);
  #pragma unroll
  for (int nt = 0; nt < 16; nt++) {
    v8f acc = {};
    #pragma unroll
    for (int ks = 0; ks < 5; ks++) {
      const unsigned short* bp = w1L + (((size_t)(ks * 16 + nt) * 32 + lane) << 4);
      v16bf b = make_frag(*(const uint4*)bp, *(const uint4*)(bp + 8));
      acc = wmma_bf16(a1[ks], b, acc);
    }
    const int n = nt * 16 + (lane & 15);
    const float bias = b1[L * HID + n];
    #pragma unroll
    for (int r = 0; r < 8; r++) {
      float h = fmaxf(acc[r] + bias, 0.0f);
      sH[(m0 + r + 8 * half) * HID + n] = f2bf(h);
    }
  }

  // ---- GEMM2: K = 256 = 8 k-steps, N = 64 = 4 tiles ----
  v16bf a2[8];
  #pragma unroll
  for (int ks = 0; ks < 8; ks++) {
    const unsigned short* rp = sH + mrow * HID + ks * 32 + 8 * half;
    a2[ks] = make_frag(*(const uint4*)rp, *(const uint4*)(rp + 16));
  }
  const unsigned short* w2L = w2p + (size_t)L * (256 * 64);
  #pragma unroll
  for (int nt = 0; nt < 4; nt++) {
    v8f acc = {};
    #pragma unroll
    for (int ks = 0; ks < 8; ks++) {
      const unsigned short* bp = w2L + (((size_t)(ks * 4 + nt) * 32 + lane) << 4);
      v16bf b = make_frag(*(const uint4*)bp, *(const uint4*)(bp + 8));
      acc = wmma_bf16(a2[ks], b, acc);
    }
    const int n = nt * 16 + (lane & 15);
    const float bias = b2[L * D_NODE + n];
    #pragma unroll
    for (int r = 0; r < 8; r++) {
      int tn = sCol[m0 + r + 8 * half];
      unsafeAtomicAdd(&agg[(size_t)tn * D_NODE + n], acc[r] + bias);
    }
  }
}

// ---------- node update kernel (WMMA) ----------
// block = 128 threads / 64 nodes. concat(node 64 | agg 64) -> 256 (ReLU) -> 64
__global__ void __launch_bounds__(128) upd_kernel(
    int L,
    const unsigned short* __restrict__ nodes_in,
    const float* __restrict__ agg,
    const unsigned short* __restrict__ w1p,
    const float* __restrict__ b1,
    const unsigned short* __restrict__ w2p,
    const float* __restrict__ b2,
    unsigned short* __restrict__ nodes_out) {
  __shared__ __align__(16) unsigned short sU[64 * 128];
  __shared__ __align__(16) unsigned short sH[64 * HID];

  const int t = threadIdx.x;
  const int e = t & 63;
  const int nBase = blockIdx.x * 64;
  const int gi = nBase + e;
  const bool valid = gi < N_NODES;

  if (t < 64) {
    uint4* dst = (uint4*)(sU + e * 128);
    if (valid) {
      const uint4* src = (const uint4*)(nodes_in + (size_t)gi * D_NODE);
      #pragma unroll
      for (int j = 0; j < 8; j++) dst[j] = src[j];
    } else {
      uint4 z = {0u, 0u, 0u, 0u};
      #pragma unroll
      for (int j = 0; j < 8; j++) dst[j] = z;
    }
  } else {
    unsigned short* dst = sU + e * 128 + 64;
    if (valid) {
      const float* src = agg + (size_t)gi * D_NODE;
      #pragma unroll
      for (int j = 0; j < 64; j++) dst[j] = f2bf(src[j]);
    } else {
      #pragma unroll
      for (int j = 0; j < 64; j++) dst[j] = 0;
    }
  }
  __syncthreads();

  const int lane = t & 31;
  const int half = lane >> 4;
  const int wave = t >> 5;
  const int m0 = wave * 16;
  const int mrow = m0 + (lane & 15);

  // ---- GEMM1: K = 128 = 4 k-steps, N = 256 = 16 tiles ----
  v16bf a1[4];
  #pragma unroll
  for (int ks = 0; ks < 4; ks++) {
    const unsigned short* rp = sU + mrow * 128 + ks * 32 + 8 * half;
    a1[ks] = make_frag(*(const uint4*)rp, *(const uint4*)(rp + 16));
  }
  const unsigned short* w1L = w1p + (size_t)L * (128 * 256);
  #pragma unroll
  for (int nt = 0; nt < 16; nt++) {
    v8f acc = {};
    #pragma unroll
    for (int ks = 0; ks < 4; ks++) {
      const unsigned short* bp = w1L + (((size_t)(ks * 16 + nt) * 32 + lane) << 4);
      v16bf b = make_frag(*(const uint4*)bp, *(const uint4*)(bp + 8));
      acc = wmma_bf16(a1[ks], b, acc);
    }
    const int n = nt * 16 + (lane & 15);
    const float bias = b1[L * HID + n];
    #pragma unroll
    for (int r = 0; r < 8; r++) {
      float h = fmaxf(acc[r] + bias, 0.0f);
      sH[(m0 + r + 8 * half) * HID + n] = f2bf(h);
    }
  }

  // ---- GEMM2: K = 256 = 8 k-steps, N = 64 = 4 tiles ----
  v16bf a2[8];
  #pragma unroll
  for (int ks = 0; ks < 8; ks++) {
    const unsigned short* rp = sH + mrow * HID + ks * 32 + 8 * half;
    a2[ks] = make_frag(*(const uint4*)rp, *(const uint4*)(rp + 16));
  }
  const unsigned short* w2L = w2p + (size_t)L * (256 * 64);
  #pragma unroll
  for (int nt = 0; nt < 4; nt++) {
    v8f acc = {};
    #pragma unroll
    for (int ks = 0; ks < 8; ks++) {
      const unsigned short* bp = w2L + (((size_t)(ks * 4 + nt) * 32 + lane) << 4);
      v16bf b = make_frag(*(const uint4*)bp, *(const uint4*)(bp + 8));
      acc = wmma_bf16(a2[ks], b, acc);
    }
    const int n = nt * 16 + (lane & 15);
    const float bias = b2[L * D_NODE + n];
    #pragma unroll
    for (int r = 0; r < 8; r++) {
      int gm = nBase + m0 + r + 8 * half;
      if (gm < N_NODES) nodes_out[(size_t)gm * D_NODE + n] = f2bf(acc[r] + bias);
    }
  }
}

// ---------- heads (tiny, VALU) ----------

__global__ void glob_embed_kernel(const float* __restrict__ xg,
                                  const float* __restrict__ w,
                                  const float* __restrict__ b,
                                  float* __restrict__ g) {
  int n = threadIdx.x;  // HID threads
  float s = b[n];
  #pragma unroll 4
  for (int k = 0; k < IN_GLOB; k++) s += xg[k] * w[k * HID + n];
  g[n] = s;
}

__global__ void head_hidden_kernel(const float* __restrict__ g,
                                   const float* __restrict__ pw1, const float* __restrict__ pb1,
                                   const float* __restrict__ vw1, const float* __restrict__ vb1,
                                   float* __restrict__ ph, float* __restrict__ vh) {
  int t = blockIdx.x * blockDim.x + threadIdx.x;
  if (t < HID) {
    float s = pb1[t];
    for (int k = 0; k < HID; k++) s += g[k] * pw1[k * HID + t];
    ph[t] = fmaxf(s, 0.0f);
  } else if (t < HID + HID / 2) {
    int v = t - HID;
    float s = vb1[v];
    for (int k = 0; k < HID; k++) s += g[k] * vw1[k * (HID / 2) + v];
    vh[v] = fmaxf(s, 0.0f);
  }
}

__global__ void head_out_kernel(const float* __restrict__ ph,
                                const float* __restrict__ pw2, const float* __restrict__ pb2,
                                const float* __restrict__ vh,
                                const float* __restrict__ vw2, const float* __restrict__ vb2,
                                float* __restrict__ out) {
  int t = blockIdx.x * blockDim.x + threadIdx.x;
  if (t < POLICY_DIM) {
    float s = pb2[t];
    for (int k = 0; k < HID; k++) s += ph[k] * pw2[(size_t)k * POLICY_DIM + t];
    out[t] = s;
  } else if (t == POLICY_DIM) {
    float s = vb2[0];
    for (int k = 0; k < HID / 2; k++) s += vh[k] * vw2[k];
    out[POLICY_DIM] = tanhf(s);
  }
}

// ---------- host launch ----------

extern "C" void kernel_launch(void* const* d_in, const int* in_sizes, int n_in,
                              void* d_out, int out_size, void* d_ws, size_t ws_size,
                              hipStream_t stream) {
  (void)in_sizes; (void)n_in; (void)out_size; (void)ws_size;

  const float* x_nodes = (const float*)d_in[0];
  const int*   eidx    = (const int*)  d_in[1];
  const float* x_edges = (const float*)d_in[2];
  const float* x_glob  = (const float*)d_in[3];
  const float* node_w  = (const float*)d_in[4];
  const float* node_b  = (const float*)d_in[5];
  const float* edge_w  = (const float*)d_in[6];
  const float* edge_b  = (const float*)d_in[7];
  const float* glob_w  = (const float*)d_in[8];
  const float* glob_b  = (const float*)d_in[9];
  const float* msg_w1  = (const float*)d_in[10];
  const float* msg_b1  = (const float*)d_in[11];
  const float* msg_w2  = (const float*)d_in[12];
  const float* msg_b2  = (const float*)d_in[13];
  const float* upd_w1  = (const float*)d_in[14];
  const float* upd_b1  = (const float*)d_in[15];
  const float* upd_w2  = (const float*)d_in[16];
  const float* upd_b2  = (const float*)d_in[17];
  const float* pol_w1  = (const float*)d_in[18];
  const float* pol_b1  = (const float*)d_in[19];
  const float* pol_w2  = (const float*)d_in[20];
  const float* pol_b2  = (const float*)d_in[21];
  const float* val_w1  = (const float*)d_in[22];
  const float* val_b1  = (const float*)d_in[23];
  const float* val_w2  = (const float*)d_in[24];
  const float* val_b2  = (const float*)d_in[25];

  char* ws = (char*)d_ws;
  size_t off = 0;
  auto take = [&](size_t bytes) -> char* {
    char* p = ws + off;
    off = (off + bytes + 255) & ~(size_t)255;
    return p;
  };
  unsigned short* nodesA = (unsigned short*)take((size_t)N_NODES * D_NODE * 2);
  unsigned short* nodesB = (unsigned short*)take((size_t)N_NODES * D_NODE * 2);
  unsigned short* edgesB = (unsigned short*)take((size_t)N_EDGES * D_EDGE * 2);
  float*          agg    = (float*)take((size_t)N_NODES * D_NODE * 4);
  unsigned short* pw_m1  = (unsigned short*)take((size_t)N_LAYERS * 160 * 256 * 2);
  unsigned short* pw_m2  = (unsigned short*)take((size_t)N_LAYERS * 256 * 64 * 2);
  unsigned short* pw_u1  = (unsigned short*)take((size_t)N_LAYERS * 128 * 256 * 2);
  unsigned short* pw_u2  = (unsigned short*)take((size_t)N_LAYERS * 256 * 64 * 2);
  float* g  = (float*)take(HID * 4);
  float* ph = (float*)take(HID * 4);
  float* vh = (float*)take((HID / 2) * 4);

  // pack weights into WMMA B-fragment order (bf16), per layer
  for (int l = 0; l < N_LAYERS; l++) {
    pack_w_kernel<<<(160 * 256 + 255) / 256, 256, 0, stream>>>(
        msg_w1 + (size_t)l * 160 * 256, pw_m1 + (size_t)l * 160 * 256, 160, 256);
    pack_w_kernel<<<(256 * 64 + 255) / 256, 256, 0, stream>>>(
        msg_w2 + (size_t)l * 256 * 64, pw_m2 + (size_t)l * 256 * 64, 256, 64);
    pack_w_kernel<<<(128 * 256 + 255) / 256, 256, 0, stream>>>(
        upd_w1 + (size_t)l * 128 * 256, pw_u1 + (size_t)l * 128 * 256, 128, 256);
    pack_w_kernel<<<(256 * 64 + 255) / 256, 256, 0, stream>>>(
        upd_w2 + (size_t)l * 256 * 64, pw_u2 + (size_t)l * 256 * 64, 256, 64);
  }

  embed_nodes_kernel<<<(N_NODES * D_NODE + 255) / 256, 256, 0, stream>>>(
      x_nodes, node_w, node_b, nodesA);
  embed_edges_kernel<<<(int)(((long long)N_EDGES * D_EDGE + 255) / 256), 256, 0, stream>>>(
      x_edges, edge_w, edge_b, edgesB);
  glob_embed_kernel<<<1, HID, 0, stream>>>(x_glob, glob_w, glob_b, g);

  unsigned short* nin = nodesA;
  unsigned short* nout = nodesB;
  for (int l = 0; l < N_LAYERS; l++) {
    zero_f32_kernel<<<(N_NODES * D_NODE + 255) / 256, 256, 0, stream>>>(agg, N_NODES * D_NODE);
    msg_kernel<<<N_EDGES / 64, 128, 0, stream>>>(
        l, eidx, nin, edgesB, pw_m1, msg_b1, pw_m2, msg_b2, agg);
    upd_kernel<<<(N_NODES + 63) / 64, 128, 0, stream>>>(
        l, nin, agg, pw_u1, upd_b1, pw_u2, upd_b2, nout);
    unsigned short* tmp = nin; nin = nout; nout = tmp;
  }

  head_hidden_kernel<<<3, 128, 0, stream>>>(g, pol_w1, pol_b1, val_w1, val_b1, ph, vh);
  head_out_kernel<<<(POLICY_DIM + 256) / 256, 256, 0, stream>>>(
      ph, pol_w2, pol_b2, vh, val_w2, val_b2, (float*)d_out);
}
